// TransformerEncoder_67534065762555
// MI455X (gfx1250) — matrix operand order repaired and checked by reading, not compiled, not run on hardware
//
#include <hip/hip_runtime.h>

#define TPD   544           // padded sequence (513 -> 544, multiple of 32)
#define EMBD  256
#define NHEADS 8
#define LAYERS 6
#define FFD   1024
#define BATCH 32
#define TIN   512
#define INDIM 64
#define BT    (BATCH*TPD)   // 17408 rows

typedef __attribute__((ext_vector_type(16))) __bf16        v16bf;
typedef __attribute__((ext_vector_type(8)))  float         v8f;
typedef __attribute__((ext_vector_type(4)))  unsigned int  v4u;
typedef __attribute__((ext_vector_type(4)))  float         v4f;

struct FragBF { union { v16bf v; v4u q[2]; }; };

__device__ __forceinline__ unsigned short f2bf(float f) {
  unsigned int u = __float_as_uint(f);
  u += 0x7fffu + ((u >> 16) & 1u);          // round-to-nearest-even
  return (unsigned short)(u >> 16);
}

#define WMMA_BF16(a, b, c) \
  __builtin_amdgcn_wmma_f32_16x16x32_bf16(false, (a), false, (b), (short)0, (c), false, false)

// ---------------------------------------------------------------------------
// Generic bf16 WMMA GEMM:  C = A[M,K] * B^T  where Bm is stored [N,K] row-major
// so each lane loads contiguous K-chunks for both operands.
// Wave tile 64x64 (4x4 WMMA tiles): 8 fragment loads feed 16 WMMAs per K-step
// (2x the operand reuse of a 32x32 tile -> half the L0/L1 traffic per FLOP).
// Block = 8 waves arranged 2(M) x 4(N) -> 128x256 block tile.
// Optional batch dim z decomposed as (bb, hh) with per-axis element offsets.
// Epilogue modes:
//  0: outF[cOff+m*ldc+n] = v + bias[n]                (final projection)
//  1: outF[m*ldc+n]      = v + bias[n] + resid[m*ldc+n]
//  2: outB[cOff+m*ldc+n] = bf16(v)
//  3: outB[m*ldc+n]      = bf16(relu(v + bias[n]))
//  4: K/V store transposed per head: outB[((b*8+h)*256+f)*TPD + t] = bf16(v)
//  5: embed: outF[(b*TPD + t + 1)*256 + n] = v + bias[n]   (m = b*512 + t)
//  6: outB[cOff+n*ldc+m] = bf16(v)                    (transposed store, KV)
// ---------------------------------------------------------------------------
__global__ __launch_bounds__(256)
void wmma_gemm(const unsigned short* __restrict__ A, int lda, long aOB, long aOH,
               const unsigned short* __restrict__ Bm, int ldb, long bOB, long bOH,
               int M, int N, int K, int Hh, int mode, int ldc,
               const float* __restrict__ bias, const float* __restrict__ resid,
               float* __restrict__ outF, unsigned short* __restrict__ outB,
               long cOB, long cOH)
{
  int z  = blockIdx.z;
  int bb = z / Hh, hh = z - bb * Hh;
  const unsigned short* Az = A + (long)bb * aOB + (long)hh * aOH;
  const unsigned short* Bz = Bm + (long)bb * bOB + (long)hh * bOH;
  long cOff = (long)bb * cOB + (long)hh * cOH;

  int wave = threadIdx.x >> 5, lane = threadIdx.x & 31;
  int m0 = blockIdx.x * 128 + (wave & 1) * 64;
  int n0 = blockIdx.y * 256 + (wave >> 1) * 64;
  if (m0 >= M || n0 >= N) return;   // whole-wave guard (strip fully out of range)

  int l15 = lane & 15, hsel = lane >> 4;

  // A fragment (16x32 bf16): lane row = base+l15; lanes 0-15 take K[0:8],K[16:24],
  // lanes 16-31 take K[8:16],K[24:32]  -> base shifted by hsel*8.
  // B fragment (32x16): lane column = base+l15; lanes 0-15 take K[0:8],K[16:24],
  // lanes 16-31 take K[8:16],K[24:32] -> base shifted by hsel*8.
  const unsigned short* aR[4];
  const unsigned short* bR[4];
  #pragma unroll
  for (int i = 0; i < 4; i++)
    aR[i] = Az + (long)(m0 + i * 16 + l15) * lda + hsel * 8;
  #pragma unroll
  for (int j = 0; j < 4; j++)
    bR[j] = Bz + (long)(n0 + j * 16 + l15) * ldb + hsel * 8;

  v8f acc[4][4] = {};
  for (int k0 = 0; k0 < K; k0 += 32) {
    FragBF af[4], bg[4];
    #pragma unroll
    for (int i = 0; i < 4; i++) {
      af[i].q[0] = *(const v4u*)(aR[i] + k0);
      af[i].q[1] = *(const v4u*)(aR[i] + k0 + 16);
    }
    #pragma unroll
    for (int j = 0; j < 4; j++) {
      bg[j].q[0] = *(const v4u*)(bR[j] + k0);
      bg[j].q[1] = *(const v4u*)(bR[j] + k0 + 16);
    }
    #pragma unroll
    for (int i = 0; i < 4; i++)
      #pragma unroll
      for (int j = 0; j < 4; j++)
        acc[i][j] = WMMA_BF16(af[i].v, bg[j].v, acc[i][j]);
  }

  #pragma unroll
  for (int i = 0; i < 4; i++) {
    int mB = m0 + i * 16 + hsel * 8;
    #pragma unroll
    for (int j = 0; j < 4; j++) {
      int n = n0 + j * 16 + l15;
      #pragma unroll
      for (int r = 0; r < 8; r++) {
        int m = mB + r;
        if (m >= M) continue;     // ragged M (attention M=544, final M=32)
        float v = acc[i][j][r];
        if (mode == 0) {
          outF[cOff + (long)m * ldc + n] = v + bias[n];
        } else if (mode == 1) {
          long o1 = (long)m * ldc + n;
          outF[o1] = v + bias[n] + resid[o1];
        } else if (mode == 2) {
          outB[cOff + (long)m * ldc + n] = f2bf(v);
        } else if (mode == 3) {
          float t1 = v + bias[n];
          outB[(long)m * ldc + n] = f2bf(t1 > 0.f ? t1 : 0.f);
        } else if (mode == 4) {
          int b_ = m / TPD, t_ = m - b_ * TPD;
          int h_ = n >> 8, f_ = n & 255;
          outB[((long)((b_ * NHEADS + h_) * EMBD + f_)) * TPD + t_] = f2bf(v);
        } else if (mode == 5) {
          int b_ = m >> 9, t_ = m & 511;
          outF[((long)(b_ * TPD + t_ + 1)) * EMBD + n] = v + bias[n];
        } else {  // mode 6
          outB[cOff + (long)n * ldc + m] = f2bf(v);
        }
      }
    }
  }
}

// ---------------------------------------------------------------------------
// LayerNorm: one wave per 256-wide row (8 rows / 256-thread block).
// Writes fp32 H and bf16 mirror Hbf; force-zeroes pad rows (t >= 513) so the
// softmax-free attention never mixes pad rows into real outputs.
// ---------------------------------------------------------------------------
__global__ __launch_bounds__(256)
void ln_kernel(const float* __restrict__ S, const float* __restrict__ g,
               const float* __restrict__ be, float* __restrict__ H,
               unsigned short* __restrict__ Hbf)
{
  int wave = threadIdx.x >> 5, lane = threadIdx.x & 31;
  long row = (long)blockIdx.x * 8 + wave;
  int  t   = (int)(row % TPD);
  const float* p = S + row * EMBD + lane * 8;
  v4f u0 = *(const v4f*)p, u1 = *(const v4f*)(p + 4);
  float x[8] = {u0.x, u0.y, u0.z, u0.w, u1.x, u1.y, u1.z, u1.w};

  float s = 0.f;
  #pragma unroll
  for (int r = 0; r < 8; r++) s += x[r];
  for (int o1 = 16; o1; o1 >>= 1) s += __shfl_xor(s, o1, 32);
  float mean = s * (1.f / EMBD);

  float q = 0.f;
  #pragma unroll
  for (int r = 0; r < 8; r++) { float d = x[r] - mean; q += d * d; }
  for (int o1 = 16; o1; o1 >>= 1) q += __shfl_xor(q, o1, 32);
  float rs = rsqrtf(q * (1.f / EMBD) + 1e-5f);

  float* oH = H + row * EMBD + lane * 8;
  unsigned short* oB = Hbf + row * EMBD + lane * 8;
  if (t >= 513) {
    #pragma unroll
    for (int r = 0; r < 8; r++) { oH[r] = 0.f; oB[r] = 0; }
  } else {
    int e = lane * 8;
    #pragma unroll
    for (int r = 0; r < 8; r++) {
      float y = (x[r] - mean) * rs * g[e + r] + be[e + r];
      oH[r] = y; oB[r] = f2bf(y);
    }
  }
}

// Transpose + convert fp32 [K,N] -> bf16 [N,K] (scale folded for q/k).
__global__ void wtrans_kernel(const float* __restrict__ in, unsigned short* __restrict__ o,
                              int K, int N, float sc)
{
  int idx = blockIdx.x * 256 + threadIdx.x;
  int n = idx / K, k = idx - n * K;
  o[idx] = f2bf(in[(long)k * N + n] * sc);
}

__global__ void cvt_kernel(const float* __restrict__ in, unsigned short* __restrict__ o, long n)
{
  long i = (long)blockIdx.x * 256 + threadIdx.x;
  if (i < n) o[i] = f2bf(in[i]);
}

// Write CLS row (t=0) and zero the pad rows (t=513..543) of H for each batch.
__global__ void cls_pad_kernel(const float* __restrict__ cls, float* __restrict__ H)
{
  int b = blockIdx.x, c = threadIdx.x;
  H[((long)b * TPD) * EMBD + c] = cls[c];
  for (int t = 513; t < TPD; ++t)
    H[((long)(b * TPD + t)) * EMBD + c] = 0.f;
}

// ---------------------------------------------------------------------------
extern "C" void kernel_launch(void* const* d_in, const int* in_sizes, int n_in,
                              void* d_out, int out_size, void* d_ws, size_t ws_size,
                              hipStream_t stream)
{
  (void)in_sizes; (void)n_in; (void)out_size; (void)ws_size;
  const float* x     = (const float*)d_in[0];
  const float* W_emb = (const float*)d_in[1];
  const float* b_emb = (const float*)d_in[2];
  const float* cls   = (const float*)d_in[3];
  const float* Wq    = (const float*)d_in[4];
  const float* Wk    = (const float*)d_in[5];
  const float* Wv    = (const float*)d_in[6];
  const float* Wu    = (const float*)d_in[7];
  const float* bu    = (const float*)d_in[8];
  const float* g1    = (const float*)d_in[9];
  const float* be1   = (const float*)d_in[10];
  const float* W1    = (const float*)d_in[11];
  const float* b1v   = (const float*)d_in[12];
  const float* W2    = (const float*)d_in[13];
  const float* b2v   = (const float*)d_in[14];
  const float* g2    = (const float*)d_in[15];
  const float* be2   = (const float*)d_in[16];
  const float* Wp    = (const float*)d_in[17];
  const float* bp    = (const float*)d_in[18];
  float* out = (float*)d_out;

  // ---- workspace carve (~325 MB) ----
  char* w = (char*)d_ws;
  auto carveF = [&](size_t n) { float* p = (float*)w; w += n * 4; return p; };
  auto carveH = [&](size_t n) { unsigned short* p = (unsigned short*)w; w += n * 2; return p; };
  float*          H    = carveF((size_t)BT * EMBD);
  float*          S    = carveF((size_t)BT * EMBD);
  unsigned short* Hbf  = carveH((size_t)BT * EMBD);
  unsigned short* Qbf  = carveH((size_t)BT * 2048);                 // also FF buffer
  unsigned short* Kt   = carveH((size_t)BT * 2048);                 // also Att buffer
  unsigned short* Vt   = carveH((size_t)BT * 2048);
  unsigned short* KVt  = carveH((size_t)BATCH * NHEADS * EMBD * EMBD);
  unsigned short* Xbf  = carveH((size_t)BATCH * TIN * INDIM);
  unsigned short* WembT= carveH((size_t)INDIM * EMBD);
  unsigned short* WqkvT= carveH((size_t)LAYERS * 3 * 2048 * 256);
  unsigned short* WuT  = carveH((size_t)LAYERS * 2048 * 256);
  unsigned short* W1T  = carveH((size_t)LAYERS * 256 * 1024);
  unsigned short* W2T  = carveH((size_t)LAYERS * 1024 * 256);
  unsigned short* WpT  = carveH((size_t)256 * 256);
  unsigned short* Att  = Kt;    // alias: K-slices consumed by KV GEMM before Att written
  unsigned short* Ff   = Qbf;   // alias: Q consumed by ATT GEMM before FF written

  auto wt = [&](const float* in, unsigned short* o, int K, int N, float sc) {
    wtrans_kernel<<<dim3((unsigned)((size_t)K * N / 256)), 256, 0, stream>>>(in, o, K, N, sc);
  };
  auto gemm = [&](const unsigned short* A, int lda, long aOB, long aOH,
                  const unsigned short* Bm, int ldb, long bOB, long bOH,
                  int M, int N, int K, int Z, int Hh, int mode, int ldcc,
                  const float* bias, const float* resid, float* oF, unsigned short* oB,
                  long cOB, long cOH) {
    dim3 g((M + 127) / 128, (N + 255) / 256, Z);
    wmma_gemm<<<g, 256, 0, stream>>>(A, lda, aOB, aOH, Bm, ldb, bOB, bOH,
                                     M, N, K, Hh, mode, ldcc, bias, resid, oF, oB, cOB, cOH);
  };

  const float qkScale = 0.25f;  // 256^-0.25, folded into Wq/Wk (exact in bf16)

  // ---- weight prep: transpose+convert to bf16 [N,K] ----
  wt(W_emb, WembT, INDIM, EMBD, 1.f);
  for (int l = 0; l < LAYERS; l++) {
    wt(Wq + (size_t)l * 524288, WqkvT + (size_t)(l * 3 + 0) * 524288, 256, 2048, qkScale);
    wt(Wk + (size_t)l * 524288, WqkvT + (size_t)(l * 3 + 1) * 524288, 256, 2048, qkScale);
    wt(Wv + (size_t)l * 524288, WqkvT + (size_t)(l * 3 + 2) * 524288, 256, 2048, 1.f);
    wt(Wu + (size_t)l * 524288, WuT + (size_t)l * 524288, 2048, 256, 1.f);
    wt(W1 + (size_t)l * 262144, W1T + (size_t)l * 262144, 256, 1024, 1.f);
    wt(W2 + (size_t)l * 262144, W2T + (size_t)l * 262144, 1024, 256, 1.f);
  }
  wt(Wp, WpT, 256, 256, 1.f);

  // ---- embedding ----
  cvt_kernel<<<dim3((unsigned)((size_t)BATCH * TIN * INDIM / 256)), 256, 0, stream>>>(
      x, Xbf, (long)BATCH * TIN * INDIM);
  cls_pad_kernel<<<BATCH, 256, 0, stream>>>(cls, H);
  // tok = x @ W_emb + b_emb into rows t=1..512 (mode 5)
  gemm(Xbf, INDIM, 0, 0, WembT, INDIM, 0, 0, BATCH * TIN, EMBD, INDIM, 1, 1, 5, EMBD,
       b_emb, nullptr, H, nullptr, 0, 0);
  cvt_kernel<<<dim3((unsigned)((size_t)BT * EMBD / 256)), 256, 0, stream>>>(
      H, Hbf, (long)BT * EMBD);

  // ---- transformer layers ----
  for (int l = 0; l < LAYERS; l++) {
    const unsigned short* WqTl = WqkvT + (size_t)(l * 3 + 0) * 524288;
    const unsigned short* WkTl = WqkvT + (size_t)(l * 3 + 1) * 524288;
    const unsigned short* WvTl = WqkvT + (size_t)(l * 3 + 2) * 524288;

    // q (scaled) -> Qbf [BT, 2048]
    gemm(Hbf, EMBD, 0, 0, WqTl, EMBD, 0, 0, BT, 2048, EMBD, 1, 1, 2, 2048,
         nullptr, nullptr, nullptr, Qbf, 0, 0);
    // k (scaled) -> Kt transposed per head [b,h,f,t]
    gemm(Hbf, EMBD, 0, 0, WkTl, EMBD, 0, 0, BT, 2048, EMBD, 1, 1, 4, 0,
         nullptr, nullptr, nullptr, Kt, 0, 0);
    // v -> Vt transposed per head
    gemm(Hbf, EMBD, 0, 0, WvTl, EMBD, 0, 0, BT, 2048, EMBD, 1, 1, 4, 0,
         nullptr, nullptr, nullptr, Vt, 0, 0);

    // KV[b,h] = k^T v  (256x256, K=TPD), stored transposed -> KVt
    gemm(Kt, TPD, 8L * EMBD * TPD, (long)EMBD * TPD,
         Vt, TPD, 8L * EMBD * TPD, (long)EMBD * TPD,
         EMBD, EMBD, TPD, BATCH * NHEADS, NHEADS, 6, EMBD,
         nullptr, nullptr, nullptr, KVt, 8L * EMBD * EMBD, (long)EMBD * EMBD);

    // att[b,h] = q @ KV  -> Att [BT, 2048]
    gemm(Qbf, 2048, (long)TPD * 2048, 256,
         KVt, EMBD, 8L * EMBD * EMBD, (long)EMBD * EMBD,
         TPD, EMBD, EMBD, BATCH * NHEADS, NHEADS, 2, 2048,
         nullptr, nullptr, nullptr, Att, (long)TPD * 2048, 256);

    // S = att @ Wu + bu + H
    gemm(Att, 2048, 0, 0, WuT + (size_t)l * 524288, 2048, 0, 0, BT, EMBD, 2048, 1, 1, 1, EMBD,
         bu + (size_t)l * 256, H, S, nullptr, 0, 0);
    ln_kernel<<<BT / 8, 256, 0, stream>>>(S, g1 + (size_t)l * 256, be1 + (size_t)l * 256, H, Hbf);

    // FF1: relu(H @ W1 + b1) -> Ff bf16
    gemm(Hbf, EMBD, 0, 0, W1T + (size_t)l * 262144, EMBD, 0, 0, BT, FFD, EMBD, 1, 1, 3, FFD,
         b1v + (size_t)l * 1024, nullptr, nullptr, Ff, 0, 0);
    // FF2: S = Ff @ W2 + b2 + H
    gemm(Ff, FFD, 0, 0, W2T + (size_t)l * 262144, FFD, 0, 0, BT, EMBD, FFD, 1, 1, 1, EMBD,
         b2v + (size_t)l * 256, H, S, nullptr, 0, 0);
    ln_kernel<<<BT / 8, 256, 0, stream>>>(S, g2 + (size_t)l * 256, be2 + (size_t)l * 256, H, Hbf);
  }

  // ---- final: out[b] = H[b, t=0] @ Wp + bp  (lda strides over batches) ----
  gemm(Hbf, TPD * EMBD, 0, 0, WpT, EMBD, 0, 0, BATCH, EMBD, EMBD, 1, 1, 0, EMBD,
       bp, nullptr, out, nullptr, 0, 0);
}
